// GNNTransformerEncoder_48344151884176
// MI455X (gfx1250) — matrix operand order repaired and checked by reading, not compiled, not run on hardware
//
#include <hip/hip_runtime.h>
#include <hip/hip_bf16.h>
#include <cstdint>

typedef __bf16 bf16;
typedef __attribute__((ext_vector_type(16))) __bf16 v16bf;
typedef __attribute__((ext_vector_type(8)))  __bf16 v8bf;
typedef __attribute__((ext_vector_type(8)))  float  v8f;

#define BSZ 16
#define SEQ 512
#define DIM 256
#define NH  4
#define NL  6
#define DHD 64
#define NND (BSZ*SEQ)      /* 8192 nodes */
#define NE  (NND*16)       /* 131072 edges */

// ---------------------------------------------------------------------------
// Generic NT WMMA GEMM: C[m,n] = sum_k A[m,k]*B[n,k], bf16 in, fp32 acc.
// One wave computes a 64x64 tile. Epilogue: *scale, +bias[n], +bmat[m,n],
// activation (0 none / 1 relu / 2 silu), +residual, dual fp32/bf16 outputs.
// Batch offsets use (bt/div)*stride1 + (bt%div)*stride2 to support the
// attention head-interleave on output and per-graph attention bias.
// ---------------------------------------------------------------------------
__global__ __launch_bounds__(32) void wmma_gemm_nt(
    const bf16* __restrict__ A, long long sA, int lda,
    const bf16* __restrict__ B, long long sB, int ldb,
    float* Cf, long long sC1, long long sC2, int divC, int ldc,
    bf16* Cb, long long sD1, long long sD2, int divD, int ldcb,
    const float* __restrict__ bias,
    const float* __restrict__ bmat, long long sM1, int divM, int ldm,
    float scale,
    const float* __restrict__ res, int res_mod, int ldr,
    int act, int K)
{
    const int lane   = threadIdx.x & 31;
    const int lanehi = lane >> 4;
    const int lanelo = lane & 15;
    const int nBase  = blockIdx.x * 64;
    const int mBase  = blockIdx.y * 64;
    const int bt     = blockIdx.z;

    const bf16* Ab = A + (long long)bt * sA;
    const bf16* Bb = B + (long long)bt * sB;

    v8f acc[4][4];
#pragma unroll
    for (int i = 0; i < 4; i++)
#pragma unroll
        for (int j = 0; j < 4; j++)
#pragma unroll
            for (int e = 0; e < 8; e++) acc[i][j][e] = 0.0f;

    for (int k0 = 0; k0 < K; k0 += 32) {
        v16bf af[4], bfr[4];
#pragma unroll
        for (int mi = 0; mi < 4; mi++) {
            const bf16* p = Ab + (long long)(mBase + mi*16 + lanelo) * lda
                               + k0 + lanehi*8;
            union { v16bf v; v8bf h[2]; } u;
            u.h[0] = *reinterpret_cast<const v8bf*>(p);
            u.h[1] = *reinterpret_cast<const v8bf*>(p + 16);
            af[mi] = u.v;
        }
#pragma unroll
        for (int ni = 0; ni < 4; ni++) {
            const bf16* p = Bb + (long long)(nBase + ni*16 + lanelo) * ldb
                               + k0 + lanehi*8;
            union { v16bf v; v8bf h[2]; } u;
            u.h[0] = *reinterpret_cast<const v8bf*>(p);
            u.h[1] = *reinterpret_cast<const v8bf*>(p + 16);
            bfr[ni] = u.v;
        }
#pragma unroll
        for (int mi = 0; mi < 4; mi++)
#pragma unroll
            for (int ni = 0; ni < 4; ni++)
                acc[mi][ni] = __builtin_amdgcn_wmma_f32_16x16x32_bf16(
                    false, af[mi], false, bfr[ni], (short)0, acc[mi][ni],
                    false, false);
    }

    float* Cfp = Cf ? (Cf + (long long)(bt / divC) * sC1
                          + (long long)(bt % divC) * sC2) : nullptr;
    bf16*  Cbp = Cb ? (Cb + (long long)(bt / divD) * sD1
                          + (long long)(bt % divD) * sD2) : nullptr;
    const float* bm = bmat ? (bmat + (long long)(bt / divM) * sM1) : nullptr;

#pragma unroll
    for (int mi = 0; mi < 4; mi++)
#pragma unroll
        for (int ni = 0; ni < 4; ni++) {
            const int n = nBase + ni*16 + lanelo;
#pragma unroll
            for (int e = 0; e < 8; e++) {
                const int m = mBase + mi*16 + lanehi*8 + e;
                float v = acc[mi][ni][e] * scale;
                if (bm)   v += bm[(long long)m * ldm + n];
                if (bias) v += bias[n];
                if (act == 1)      v = fmaxf(v, 0.0f);
                else if (act == 2) v = v / (1.0f + __expf(-v));
                if (res) {
                    const int rr = res_mod ? (m % res_mod) : m;
                    v += res[(long long)rr * ldr + n];
                }
                if (Cfp) Cfp[(long long)m * ldc  + n] = v;
                if (Cbp) Cbp[(long long)m * ldcb + n] = (bf16)v;
            }
        }
}

// ---------------------------------------------------------------------------
// Small utility kernels
// ---------------------------------------------------------------------------
__global__ __launch_bounds__(256) void f2bf_kernel(const float* __restrict__ s,
                                                   bf16* __restrict__ d,
                                                   long long n)
{
    long long i = (long long)blockIdx.x * 256 + threadIdx.x;
    if (i < n) d[i] = (bf16)s[i];
}

__global__ __launch_bounds__(256) void fill_f32(float* __restrict__ p, float v,
                                                long long n)
{
    long long i = (long long)blockIdx.x * 256 + threadIdx.x;
    if (i < n) p[i] = v;
}

// LayerNorm over rows of 256, optional bf16 and fp32 outputs.
__global__ __launch_bounds__(256) void ln_kernel(const float* __restrict__ x,
    const float* __restrict__ w, const float* __restrict__ b,
    bf16* __restrict__ outB, float* __restrict__ outF)
{
    __shared__ float red[256];
    const int r = blockIdx.x, t = threadIdx.x;
    const float v = x[(long long)r * 256 + t];
    red[t] = v; __syncthreads();
    for (int s = 128; s > 0; s >>= 1) {
        if (t < s) red[t] += red[t + s];
        __syncthreads();
    }
    const float mu = red[0] * (1.0f / 256.0f);
    __syncthreads();
    const float dlt = v - mu;
    red[t] = dlt * dlt; __syncthreads();
    for (int s = 128; s > 0; s >>= 1) {
        if (t < s) red[t] += red[t + s];
        __syncthreads();
    }
    const float var = red[0] * (1.0f / 256.0f);
    const float y = dlt * rsqrtf(var + 1e-5f) * w[t] + b[t];
    if (outB) outB[(long long)r * 256 + t] = (bf16)y;
    if (outF) outF[(long long)r * 256 + t] = y;
}

// Softmax over rows of 512, fp32 in -> bf16 out.
__global__ __launch_bounds__(256) void softmax512(const float* __restrict__ Sc,
                                                  bf16* __restrict__ P)
{
    __shared__ float red[256];
    const long long r = blockIdx.x;
    const int t = threadIdx.x;
    const float a = Sc[r * 512 + t];
    const float b = Sc[r * 512 + 256 + t];
    red[t] = fmaxf(a, b); __syncthreads();
    for (int s = 128; s > 0; s >>= 1) {
        if (t < s) red[t] = fmaxf(red[t], red[t + s]);
        __syncthreads();
    }
    const float mx = red[0];
    __syncthreads();
    const float ea = __expf(a - mx), eb = __expf(b - mx);
    red[t] = ea + eb; __syncthreads();
    for (int s = 128; s > 0; s >>= 1) {
        if (t < s) red[t] += red[t + s];
        __syncthreads();
    }
    const float inv = 1.0f / red[0];
    P[r * 512 + t]       = (bf16)(ea * inv);
    P[r * 512 + 256 + t] = (bf16)(eb * inv);
}

// qkv fp32 [NND,768] -> Q,K bf16 [b,h,s,dh] and V^T bf16 [b,h,dh,s]
__global__ __launch_bounds__(256) void pack_qkv(const float* __restrict__ qkv,
    bf16* __restrict__ Q, bf16* __restrict__ Kk, bf16* __restrict__ Vt)
{
    const int tid = blockIdx.x * 256 + threadIdx.x;   // < NND*DIM
    const int n  = tid >> 8;
    const int dc = tid & 255;
    const int h  = dc >> 6, dh = dc & 63;
    const int b  = n >> 9,  s  = n & 511;
    const long long base = (long long)n * 768;
    const float q = qkv[base + dc];
    const float k = qkv[base + 256 + dc];
    const float v = qkv[base + 512 + dc];
    const long long bh = (long long)(b * NH + h);
    Q [(bh * SEQ + s) * DHD + dh] = (bf16)q;
    Kk[(bh * SEQ + s) * DHD + dh] = (bf16)k;
    Vt[(bh * DHD + dh) * SEQ + s] = (bf16)v;
}

// ---------------------------------------------------------------------------
// GINE kernels
// ---------------------------------------------------------------------------
__global__ __launch_bounds__(256) void gine1_edge(const float* __restrict__ x,
    const float* __restrict__ ea, const int* __restrict__ src,
    const int* __restrict__ dst, const float* __restrict__ ew,
    const float* __restrict__ eb, float* __restrict__ agg)
{
    const int e = blockIdx.x * 256 + threadIdx.x;
    if (e >= NE) return;
    const float a = ea[e];
    const int s = src[e], d = dst[e];
    const float m0 = fmaxf(0.0f, x[s*2 + 0] + a * ew[0] + eb[0]);
    const float m1 = fmaxf(0.0f, x[s*2 + 1] + a * ew[1] + eb[1]);
    atomicAdd(&agg[d*2 + 0], m0);
    atomicAdd(&agg[d*2 + 1], m1);
}

// per-node MLP 2 -> 64 (relu) -> 128 (relu, outer GINE relu fused)
__global__ __launch_bounds__(128) void gine1_node(const float* __restrict__ x,
    const float* __restrict__ agg, const float* __restrict__ eps1p,
    const float* __restrict__ w1, const float* __restrict__ b1,
    const float* __restrict__ w2, const float* __restrict__ b2,
    float* __restrict__ h1)
{
    const int n = blockIdx.x, t = threadIdx.x;
    const float ep = 1.0f + *eps1p;
    const float z0 = ep * x[n*2 + 0] + agg[n*2 + 0];
    const float z1 = ep * x[n*2 + 1] + agg[n*2 + 1];
    __shared__ float tt[64];
    if (t < 64)
        tt[t] = fmaxf(0.0f, w1[t*2 + 0]*z0 + w1[t*2 + 1]*z1 + b1[t]);
    __syncthreads();
    float s = b2[t];
#pragma unroll 8
    for (int j = 0; j < 64; j++) s += tt[j] * w2[t*64 + j];
    h1[(long long)n*128 + t] = fmaxf(0.0f, s);
}

__global__ __launch_bounds__(256) void gine2_edge(const float* __restrict__ h1,
    const float* __restrict__ ea, const int* __restrict__ src,
    const int* __restrict__ dst, const float* __restrict__ ew,
    const float* __restrict__ eb, float* __restrict__ agg)
{
    const long long tid = (long long)blockIdx.x * 256 + threadIdx.x;
    if (tid >= (long long)NE * 128) return;
    const int e = (int)(tid >> 7);
    const int c = (int)(tid & 127);
    const int s = src[e], d = dst[e];
    const float m = fmaxf(0.0f, h1[(long long)s*128 + c] + ea[e]*ew[c] + eb[c]);
    atomicAdd(&agg[(long long)d*128 + c], m);
}

__global__ __launch_bounds__(256) void gine2_z(const float* __restrict__ h1,
    const float* __restrict__ agg, const float* __restrict__ eps2p,
    bf16* __restrict__ z2b)
{
    const long long i = (long long)blockIdx.x * 256 + threadIdx.x;
    if (i >= (long long)NND * 128) return;
    z2b[i] = (bf16)((1.0f + *eps2p) * h1[i] + agg[i]);
}

// biasM prefilled with 2.0; dropped edges write 1.0 (symmetric)
__global__ __launch_bounds__(256) void bias_edge(const int* __restrict__ src,
    const int* __restrict__ dst, const int* __restrict__ drop,
    float* __restrict__ bm)
{
    const int e = blockIdx.x * 256 + threadIdx.x;
    if (e >= NE) return;
    if (drop[e]) {
        const int s = src[e], d = dst[e];
        const int g = s >> 9, ls = s & 511, ld = d & 511;
        bm[((long long)g << 18) + (ls << 9) + ld] = 1.0f;
        bm[((long long)g << 18) + (ld << 9) + ls] = 1.0f;
    }
}

// ---------------------------------------------------------------------------
extern "C" void kernel_launch(void* const* d_in, const int* in_sizes, int n_in,
                              void* d_out, int out_size, void* d_ws, size_t ws_size,
                              hipStream_t stream)
{
    (void)in_sizes; (void)n_in; (void)out_size; (void)ws_size;
    const float* x     = (const float*)d_in[0];
    const float* eattr = (const float*)d_in[1];
    const float* eps1  = (const float*)d_in[2];
    const float* ew1   = (const float*)d_in[3];
    const float* eb1   = (const float*)d_in[4];
    const float* n1w1  = (const float*)d_in[5];
    const float* n1b1  = (const float*)d_in[6];
    const float* n1w2  = (const float*)d_in[7];
    const float* n1b2  = (const float*)d_in[8];
    const float* eps2  = (const float*)d_in[9];
    const float* ew2   = (const float*)d_in[10];
    const float* eb2   = (const float*)d_in[11];
    const float* n2w1  = (const float*)d_in[12];
    const float* n2b1  = (const float*)d_in[13];
    const float* n2w2  = (const float*)d_in[14];
    const float* n2b2  = (const float*)d_in[15];
    const float* pos   = (const float*)d_in[16];
    const float* ipw   = (const float*)d_in[17];
    const float* ipb   = (const float*)d_in[18];
    const float* opw   = (const float*)d_in[19];
    const float* opb   = (const float*)d_in[20];
    const float* ln1w  = (const float*)d_in[21];
    const float* ln1b  = (const float*)d_in[22];
    const float* ln2w  = (const float*)d_in[23];
    const float* ln2b  = (const float*)d_in[24];
    const float* f1w   = (const float*)d_in[25];
    const float* f1b   = (const float*)d_in[26];
    const float* f2w   = (const float*)d_in[27];
    const float* f2b   = (const float*)d_in[28];
    const float* fnw   = (const float*)d_in[29];
    const float* fnb   = (const float*)d_in[30];
    const int*   eidx  = (const int*)d_in[31];
    const int*   drop  = (const int*)d_in[33];
    const int* esrc = eidx;
    const int* edst = eidx + NE;

    char* ws = (char*)d_ws;
    size_t off = 0;
    auto take = [&](size_t bytes) -> char* {
        char* p = ws + off;
        off += (bytes + 255) & ~(size_t)255;
        return p;
    };

    float* scoresF = (float*)take((size_t)BSZ*NH*SEQ*SEQ*4);   // 67MB (also GINE temps)
    char*  regP    =         take((size_t)BSZ*NH*SEQ*SEQ*2);   // 33.5MB (P, aliases qkv)
    bf16*  P       = (bf16*)regP;
    float* qkv     = (float*)regP;
    bf16* Win  = (bf16*)take((size_t)NL*3*DIM*DIM*2);
    bf16* Wout = (bf16*)take((size_t)NL*DIM*DIM*2);
    bf16* Wf1  = (bf16*)take((size_t)NL*2*DIM*DIM*2);
    bf16* Wf2  = (bf16*)take((size_t)NL*DIM*2*DIM*2);
    bf16* Wn21 = (bf16*)take((size_t)DIM*128*2);
    bf16* Wn22 = (bf16*)take((size_t)DIM*DIM*2);
    float* biasM = (float*)take((size_t)BSZ*SEQ*SEQ*4);
    float* xb  = (float*)take((size_t)NND*DIM*4);
    bf16* hnb  = (bf16*)take((size_t)NND*DIM*2);
    bf16* Qb   = (bf16*)take((size_t)BSZ*NH*SEQ*DHD*2);
    bf16* Kb   = (bf16*)take((size_t)BSZ*NH*SEQ*DHD*2);
    bf16* Vt   = (bf16*)take((size_t)BSZ*NH*SEQ*DHD*2);
    bf16* Obf  = (bf16*)take((size_t)NND*DIM*2);
    bf16* ffh  = (bf16*)take((size_t)NND*2*DIM*2);

    // GINE temporaries alias the (not-yet-used) scores region
    char* g = (char*)scoresF;
    float* agg1 = (float*)g;  g += (size_t)1 << 16;
    float* h1   = (float*)g;  g += (size_t)NND*128*4;
    float* agg2 = (float*)g;  g += (size_t)NND*128*4;
    bf16*  z2b  = (bf16*)g;   g += (size_t)NND*128*2;
    bf16*  t2b  = (bf16*)g;

    auto cdiv = [](long long a, long long b) { return (int)((a + b - 1) / b); };

    // --- weight conversion fp32 -> bf16 ---
    f2bf_kernel<<<cdiv((long long)NL*3*DIM*DIM,256),256,0,stream>>>(ipw,  Win,  (long long)NL*3*DIM*DIM);
    f2bf_kernel<<<cdiv((long long)NL*DIM*DIM,256),  256,0,stream>>>(opw,  Wout, (long long)NL*DIM*DIM);
    f2bf_kernel<<<cdiv((long long)NL*2*DIM*DIM,256),256,0,stream>>>(f1w,  Wf1,  (long long)NL*2*DIM*DIM);
    f2bf_kernel<<<cdiv((long long)NL*DIM*2*DIM,256),256,0,stream>>>(f2w,  Wf2,  (long long)NL*DIM*2*DIM);
    f2bf_kernel<<<cdiv((long long)DIM*128,256),     256,0,stream>>>(n2w1, Wn21, (long long)DIM*128);
    f2bf_kernel<<<cdiv((long long)DIM*DIM,256),     256,0,stream>>>(n2w2, Wn22, (long long)DIM*DIM);

    // --- GINE ---
    fill_f32<<<cdiv((long long)NND*2,256),  256,0,stream>>>(agg1, 0.0f, (long long)NND*2);
    fill_f32<<<cdiv((long long)NND*128,256),256,0,stream>>>(agg2, 0.0f, (long long)NND*128);
    fill_f32<<<cdiv((long long)BSZ*SEQ*SEQ,256),256,0,stream>>>(biasM, 2.0f, (long long)BSZ*SEQ*SEQ);

    gine1_edge<<<cdiv(NE,256),256,0,stream>>>(x, eattr, esrc, edst, ew1, eb1, agg1);
    gine1_node<<<NND,128,0,stream>>>(x, agg1, eps1, n1w1, n1b1, n1w2, n1b2, h1);
    gine2_edge<<<cdiv((long long)NE*128,256),256,0,stream>>>(h1, eattr, esrc, edst, ew2, eb2, agg2);
    gine2_z<<<cdiv((long long)NND*128,256),256,0,stream>>>(h1, agg2, eps2, z2b);

    // t2 = relu(z2 @ nn2_w1^T + b1)   [NND,256] K=128
    wmma_gemm_nt<<<dim3(DIM/64, NND/64, 1),32,0,stream>>>(
        z2b, 0, 128, Wn21, 0, 128,
        nullptr, 0, 0, 1, 0,
        t2b, 0, 0, 1, DIM,
        n2b1,
        nullptr, 0, 1, 0, 1.0f,
        nullptr, 0, 0,
        1, 128);
    // xb = t2 @ nn2_w2^T + b2 + pos_emb[s]   [NND,256] K=256
    wmma_gemm_nt<<<dim3(DIM/64, NND/64, 1),32,0,stream>>>(
        t2b, 0, DIM, Wn22, 0, DIM,
        xb, 0, 0, 1, DIM,
        nullptr, 0, 0, 1, 0,
        n2b2,
        nullptr, 0, 1, 0, 1.0f,
        pos, SEQ, DIM,
        0, DIM);

    bias_edge<<<cdiv(NE,256),256,0,stream>>>(esrc, edst, drop, biasM);

    // --- transformer layers ---
    for (int l = 0; l < NL; l++) {
        ln_kernel<<<NND,256,0,stream>>>(xb, ln1w + l*DIM, ln1b + l*DIM, hnb, nullptr);

        // qkv = hn @ in_proj^T + b   [NND,768] K=256
        wmma_gemm_nt<<<dim3(3*DIM/64, NND/64, 1),32,0,stream>>>(
            hnb, 0, DIM, Win + (size_t)l*3*DIM*DIM, 0, DIM,
            qkv, 0, 0, 1, 3*DIM,
            nullptr, 0, 0, 1, 0,
            ipb + l*3*DIM,
            nullptr, 0, 1, 0, 1.0f,
            nullptr, 0, 0,
            0, DIM);

        pack_qkv<<<cdiv((long long)NND*DIM,256),256,0,stream>>>(qkv, Qb, Kb, Vt);

        // scores = Q·K^T * 1/8 + biasM   batched over 64 (b,h)
        wmma_gemm_nt<<<dim3(SEQ/64, SEQ/64, BSZ*NH),32,0,stream>>>(
            Qb, (long long)SEQ*DHD, DHD, Kb, (long long)SEQ*DHD, DHD,
            scoresF, (long long)SEQ*SEQ, 0, 1, SEQ,
            nullptr, 0, 0, 1, 0,
            nullptr,
            biasM, (long long)SEQ*SEQ, NH, SEQ, 0.125f,
            nullptr, 0, 0,
            0, DHD);

        softmax512<<<BSZ*NH*SEQ,256,0,stream>>>(scoresF, P);

        // O = P @ V  -> bf16 directly into [b,s, h*64+dh] layout
        wmma_gemm_nt<<<dim3(1, SEQ/64, BSZ*NH),32,0,stream>>>(
            P, (long long)SEQ*SEQ, SEQ, Vt, (long long)SEQ*DHD, SEQ,
            nullptr, 0, 0, 1, 0,
            Obf, (long long)SEQ*DIM, DHD, NH, DIM,
            nullptr,
            nullptr, 0, 1, 0, 1.0f,
            nullptr, 0, 0,
            0, SEQ);

        // xb = xb + O @ out_proj^T + b
        wmma_gemm_nt<<<dim3(DIM/64, NND/64, 1),32,0,stream>>>(
            Obf, 0, DIM, Wout + (size_t)l*DIM*DIM, 0, DIM,
            xb, 0, 0, 1, DIM,
            nullptr, 0, 0, 1, 0,
            opb + l*DIM,
            nullptr, 0, 1, 0, 1.0f,
            xb, 0, DIM,
            0, DIM);

        ln_kernel<<<NND,256,0,stream>>>(xb, ln2w + l*DIM, ln2b + l*DIM, hnb, nullptr);

        // ffh = silu(hn @ ff_w1^T + b1)   [NND,512]
        wmma_gemm_nt<<<dim3(2*DIM/64, NND/64, 1),32,0,stream>>>(
            hnb, 0, DIM, Wf1 + (size_t)l*2*DIM*DIM, 0, DIM,
            nullptr, 0, 0, 1, 0,
            ffh, 0, 0, 1, 2*DIM,
            f1b + l*2*DIM,
            nullptr, 0, 1, 0, 1.0f,
            nullptr, 0, 0,
            2, DIM);

        // xb = xb + ffh @ ff_w2^T + b2   K=512
        wmma_gemm_nt<<<dim3(DIM/64, NND/64, 1),32,0,stream>>>(
            ffh, 0, 2*DIM, Wf2 + (size_t)l*DIM*2*DIM, 0, 2*DIM,
            xb, 0, 0, 1, DIM,
            nullptr, 0, 0, 1, 0,
            f2b + l*DIM,
            nullptr, 0, 1, 0, 1.0f,
            xb, 0, DIM,
            0, 2*DIM);
    }

    ln_kernel<<<NND,256,0,stream>>>(xb, fnw, fnb, nullptr, (float*)d_out);
}
